// Decoder_4999341932805
// MI455X (gfx1250) — matrix operand order repaired and checked by reading, not compiled, not run on hardware
//
#include <hip/hip_runtime.h>
#include <hip/hip_bf16.h>

#define BB 256      // batch
#define TT 512      // time steps
#define HH 1024     // hidden
#define NBLK 128    // persistent grid size
#define NTHR 256    // threads per block (8 waves)

typedef unsigned short bf16raw;
typedef __bf16 bf16;
typedef __attribute__((ext_vector_type(8)))  bf16  v8bf;
typedef __attribute__((ext_vector_type(16))) bf16  v16bf;
typedef __attribute__((ext_vector_type(8)))  float v8f;

__device__ __forceinline__ float sigf(float x)     { return 1.0f / (1.0f + __expf(-x)); }
__device__ __forceinline__ float tanhfast(float x) { return 1.0f - 2.0f / (__expf(2.0f * x) + 1.0f); }

// Device-scope grid barrier: counter + generation flag in global memory.
__device__ __forceinline__ void grid_barrier(unsigned* cnt, unsigned* gen) {
    __threadfence();          // flush this thread's h writes to device scope
    __syncthreads();
    if (threadIdx.x == 0) {
        unsigned g = __hip_atomic_load(gen, __ATOMIC_RELAXED, __HIP_MEMORY_SCOPE_AGENT);
        unsigned prev = __hip_atomic_fetch_add(cnt, 1u, __ATOMIC_ACQ_REL, __HIP_MEMORY_SCOPE_AGENT);
        if (prev == NBLK - 1u) {
            __hip_atomic_store(cnt, 0u, __ATOMIC_RELAXED, __HIP_MEMORY_SCOPE_AGENT);
            __hip_atomic_fetch_add(gen, 1u, __ATOMIC_RELEASE, __HIP_MEMORY_SCOPE_AGENT);
        } else {
            while (__hip_atomic_load(gen, __ATOMIC_ACQUIRE, __HIP_MEMORY_SCOPE_AGENT) == g)
                __builtin_amdgcn_s_sleep(2);
        }
    }
    __syncthreads();
    __threadfence();          // acquire: invalidate stale cached h lines
}

__global__ void init_sync_kernel(unsigned* sync) { sync[0] = 0u; sync[1] = 0u; }

// Persistent LSTM kernel.
// Block b owns batch rows [ (b>>6)*128 , +128 ) and hidden cols [ (b&63)*16 , +16 ).
// Each of the 8 waves owns a 16x16 (batch x hidden) tile; c-state lives in VGPRs.
// W_hh slice (4 gates x 16 cols x 1024 K, bf16, WMMA-B swizzled) lives in 128KB LDS.
__global__ void __launch_bounds__(NTHR) lstm_persistent(
    const float* __restrict__ y_hist, const float* __restrict__ W_ih,
    const float* __restrict__ W_hh,   const float* __restrict__ b_ih,
    const float* __restrict__ b_hh,   const float* __restrict__ h0,
    const float* __restrict__ c0,
    bf16raw* __restrict__ hb0, bf16raw* __restrict__ hb1,
    float* __restrict__ hfin, unsigned* __restrict__ sync)
{
    extern __shared__ bf16 sW[];                 // 4*32*32*16 = 65536 bf16 = 128KB

    const int tid  = threadIdx.x;
    const int lane = tid & 31;
    const int wave = tid >> 5;
    const int n0   = (blockIdx.x & 63) << 4;     // hidden-column base of block
    const int m0   = ((blockIdx.x >> 6) << 7) + (wave << 4);  // batch-row base of wave
    const int nl   = lane & 15;
    const int hi16 = lane >> 4;
    const int n    = n0 + nl;                    // this lane's hidden column

    // ---- Stage W_hh slice into LDS (bf16, swizzled for WMMA B-fragments), once ----
    for (int idx = tid; idx < 4 * 32 * 32 * 16; idx += NTHR) {
        int e  = idx & 15;
        int ls = (idx >> 4) & 31;
        int kk = (idx >> 9) & 31;
        int g  = idx >> 14;
        int nn = n0 + (ls & 15);
        int k  = (kk << 5) + ((e & 8) << 1) + ((ls >> 4) << 3) + (e & 7);
        sW[idx] = (bf16)W_hh[(size_t)(g * HH + nn) * HH + k];
    }

    // ---- Convert h0 slice to bf16 into buffer 0 ----
    {
        size_t base = (size_t)blockIdx.x * (BB * HH / NBLK);
        for (int i = tid; i < BB * HH / NBLK; i += NTHR)
            ((bf16*)hb0)[base + i] = (bf16)h0[base + i];
    }
    grid_barrier(sync, sync + 1);

    // ---- Per-lane constants (persist across all steps) ----
    float win[4], bia[4];
#pragma unroll
    for (int g = 0; g < 4; ++g) {
        int j = g * HH + n;
        win[g] = W_ih[j];                 // W_ih is (4H,1)
        bia[g] = b_ih[j] + b_hh[j];
    }
    const int brow = m0 + (hi16 << 3);    // acc vgpr v -> batch row brow+v
    float cst[8];
#pragma unroll
    for (int v = 0; v < 8; ++v) cst[v] = c0[(size_t)(brow + v) * HH + n];

    int bbase[4];
#pragma unroll
    for (int g = 0; g < 4; ++g) bbase[g] = (g * 1024 + lane) << 4;

    const int a_row = m0 + nl;            // A-fragment row for this lane
    const int koff  = hi16 ? 8 : 0;       // A-fragment K split per CDNA5 layout

    const bf16* hrd = (const bf16*)hb0;
    bf16*       hwr = (bf16*)hb1;

    for (int t = 0; t < TT; ++t) {
        const bf16* ap = hrd + (size_t)a_row * HH + koff;
        v8f acc0 = {}, acc1 = {}, acc2 = {}, acc3 = {};

#pragma unroll 4
        for (int kk = 0; kk < 32; ++kk) {
            v8bf lo = *(const v8bf*)(ap + (kk << 5));
            v8bf hi = *(const v8bf*)(ap + (kk << 5) + 16);
            v16bf a = __builtin_shufflevector(lo, hi,
                0, 1, 2, 3, 4, 5, 6, 7, 8, 9, 10, 11, 12, 13, 14, 15);
            v16bf bI = *(const v16bf*)(sW + bbase[0] + (kk << 9));
            v16bf bF = *(const v16bf*)(sW + bbase[1] + (kk << 9));
            v16bf bG = *(const v16bf*)(sW + bbase[2] + (kk << 9));
            v16bf bO = *(const v16bf*)(sW + bbase[3] + (kk << 9));
            acc0 = __builtin_amdgcn_wmma_f32_16x16x32_bf16(false, a, false, bI, (short)0, acc0, false, false);
            acc1 = __builtin_amdgcn_wmma_f32_16x16x32_bf16(false, a, false, bF, (short)0, acc1, false, false);
            acc2 = __builtin_amdgcn_wmma_f32_16x16x32_bf16(false, a, false, bG, (short)0, acc2, false, false);
            acc3 = __builtin_amdgcn_wmma_f32_16x16x32_bf16(false, a, false, bO, (short)0, acc3, false, false);
        }

        // Activations + state update (acc vgpr v -> batch row brow+v, col n)
#pragma unroll
        for (int v = 0; v < 8; ++v) {
            int br = brow + v;
            float x  = y_hist[(size_t)br * TT + t];
            float gi = acc0[v] + x * win[0] + bia[0];
            float gf = acc1[v] + x * win[1] + bia[1];
            float gg = acc2[v] + x * win[2] + bia[2];
            float go = acc3[v] + x * win[3] + bia[3];
            float c  = sigf(gf) * cst[v] + sigf(gi) * tanhfast(gg);
            float h  = sigf(go) * tanhfast(c);
            cst[v] = c;
            hwr[(size_t)br * HH + n] = (bf16)h;
            if (t == TT - 1) hfin[(size_t)br * HH + n] = h;
        }

        // swap double buffers, sync the grid
        const bf16* tmp = hrd; hrd = hwr; hwr = (bf16*)tmp;
        grid_barrier(sync, sync + 1);
    }
}

// Deterministic final FC: out[b] = dot(h_T[b,:], W_fc[0,:]) + b_fc[0]
__global__ void __launch_bounds__(256) fc_out_kernel(
    const float* __restrict__ hfin, const float* __restrict__ W_fc,
    const float* __restrict__ b_fc, float* __restrict__ out)
{
    __shared__ float red[256];
    int b = blockIdx.x, tid = threadIdx.x;
    float s = 0.f;
    for (int k = tid; k < HH; k += 256) s += hfin[(size_t)b * HH + k] * W_fc[k];
    red[tid] = s;
    __syncthreads();
    for (int off = 128; off > 0; off >>= 1) {
        if (tid < off) red[tid] += red[tid + off];
        __syncthreads();
    }
    if (tid == 0) out[b] = red[0] + b_fc[0];
}

extern "C" void kernel_launch(void* const* d_in, const int* in_sizes, int n_in,
                              void* d_out, int out_size, void* d_ws, size_t ws_size,
                              hipStream_t stream) {
    (void)in_sizes; (void)n_in; (void)out_size; (void)ws_size;
    const float* y_hist = (const float*)d_in[0];
    const float* W_ih   = (const float*)d_in[1];
    const float* W_hh   = (const float*)d_in[2];
    const float* b_ih   = (const float*)d_in[3];
    const float* b_hh   = (const float*)d_in[4];
    const float* W_fc   = (const float*)d_in[5];
    const float* b_fc   = (const float*)d_in[6];
    const float* h0     = (const float*)d_in[7];
    const float* c0     = (const float*)d_in[8];
    float* out = (float*)d_out;

    char* ws = (char*)d_ws;
    unsigned* sync   = (unsigned*)ws;                         // 2 uints (cnt, gen)
    bf16raw*  hbuf0  = (bf16raw*)(ws + 256);                  // 512 KB
    bf16raw*  hbuf1  = (bf16raw*)(ws + 256 + 524288);         // 512 KB
    float*    hfin   = (float*)  (ws + 256 + 2 * 524288);     // 1 MB

    init_sync_kernel<<<dim3(1), dim3(1), 0, stream>>>(sync);
    lstm_persistent<<<dim3(NBLK), dim3(NTHR), 131072, stream>>>(
        y_hist, W_ih, W_hh, b_ih, b_hh, h0, c0, hbuf0, hbuf1, hfin, sync);
    fc_out_kernel<<<dim3(BB), dim3(256), 0, stream>>>(hfin, W_fc, b_fc, out);
}
